// GNN_3135326126610
// MI455X (gfx1250) — compile-verified, matrix operand order checked by reading, not run on hardware
//
#include <hip/hip_runtime.h>
#include <hip/hip_bf16.h>

// ---------------------------------------------------------------------------
// GIN (3-layer) forward for MI455X (gfx1250).
// All GEMMs via V_WMMA_F32_16X16X32_F16 (f16 in, f32 accumulate).
// B=16, N=256, DIN=32, D=64, L=3, M=B*N=4096 rows.
// ---------------------------------------------------------------------------

typedef _Float16 h8  __attribute__((ext_vector_type(8)));
typedef _Float16 v16h __attribute__((ext_vector_type(16)));
typedef float    v8f  __attribute__((ext_vector_type(8)));

#define GB 16
#define GN 256
#define GD 64
#define GM 4096   // B*N

// Load 16 halves for one WMMA operand lane: [k..k+7] and [k+16..k+23]
// (ISA 16-bit A/B layout: lane half selects koff 0/8; VGPR0-3 = K[koff..],
//  VGPR4-7 = K[16+koff..]). Two aligned 16-byte loads.
__device__ __forceinline__ v16h ld16(const _Float16* p) {
    h8 lo = *(const h8*)(p);
    h8 hi = *(const h8*)(p + 16);
    return __builtin_shufflevector(lo, hi, 0,1,2,3,4,5,6,7,8,9,10,11,12,13,14,15);
}

__device__ __forceinline__ v8f wmma_f16(v16h a, v16h b, v8f c) {
    return __builtin_amdgcn_wmma_f32_16x16x32_f16(false, a, false, b,
                                                  (short)0, c, false, false);
}

// ---------------------------------------------------------------------------
// Elementwise conversion helpers
// ---------------------------------------------------------------------------
__global__ void k_f32_to_f16(const float* __restrict__ src,
                             _Float16* __restrict__ dst, int n, int clampOne) {
    for (int i = blockIdx.x * blockDim.x + threadIdx.x; i < n;
         i += gridDim.x * blockDim.x) {
        float v = src[i];
        if (clampOne) v = fminf(v, 1.0f);   // adj_ones clamp (binary adj: no-op)
        dst[i] = (_Float16)v;
    }
}

// dst[c*R + r] = src[r*C + c]  (f32 [R,C] -> f16 [C,R], i.e. N-major weights)
__global__ void k_transpose_f16(const float* __restrict__ src,
                                _Float16* __restrict__ dst, int R, int C) {
    int n = R * C;
    for (int i = blockIdx.x * blockDim.x + threadIdx.x; i < n;
         i += gridDim.x * blockDim.x) {
        int r = i / C, c = i - r * C;
        dst[c * R + r] = (_Float16)src[i];
    }
}

// msgT[b][d][j] = relu(x[b][j][d] + bond_W[l][d])   (transposed so the GEMM
// B-operand columns are contiguous in memory)
__global__ void k_message(const float* __restrict__ x,
                          const float* __restrict__ bondW, int l,
                          _Float16* __restrict__ msgT) {
    int n = GB * GD * GN;
    for (int i = blockIdx.x * blockDim.x + threadIdx.x; i < n;
         i += gridDim.x * blockDim.x) {
        int j = i & (GN - 1);
        int d = (i >> 8) & (GD - 1);
        int b = i >> 14;
        float v = x[(b * GN + j) * GD + d] + bondW[l * GD + d];
        msgT[i] = (_Float16)fmaxf(v, 0.0f);
    }
}

// ---------------------------------------------------------------------------
// Generic WMMA GEMM:  C[M,Nc](f32) = A[M,K](f16) @ Bt[Nc,K](f16)^T + bias
// One wave -> one 16x16 tile. 4 waves / block. K multiple of 32.
// ---------------------------------------------------------------------------
__global__ void k_gemm_f16(const _Float16* __restrict__ A,
                           const _Float16* __restrict__ Bt,
                           const float* __restrict__ bias,
                           float* __restrict__ C, int K, int Nc) {
    int lane = threadIdx.x & 31;
    int tile = blockIdx.x * (blockDim.x >> 5) + (threadIdx.x >> 5);
    int ntn  = Nc >> 4;
    int tn   = tile % ntn;
    int tm   = tile / ntn;
    int m    = lane & 15;
    int koff = (lane >> 4) << 3;

    const _Float16* pa = A  + (size_t)(tm * 16 + m) * K + koff;
    const _Float16* pb = Bt + (size_t)(tn * 16 + m) * K + koff;

    v8f acc = {};
    for (int k0 = 0; k0 < K; k0 += 32) {
        v16h av = ld16(pa + k0);
        v16h bv = ld16(pb + k0);
        acc = wmma_f16(av, bv, acc);
    }
    // C layout: VGPR r -> row r + 8*(lane>=16), col = lane&15
    int row0 = tm * 16 + ((lane >> 4) << 3);
    int col  = tn * 16 + (lane & 15);
    float bv = bias ? bias[col] : 0.0f;
#pragma unroll
    for (int r = 0; r < 8; ++r)
        C[(size_t)(row0 + r) * Nc + col] = acc[r] + bv;
}

// ---------------------------------------------------------------------------
// GIN aggregation GEMM + residual:
//   h16[b,i,d] = f16( adj16[b] @ msgT[b]^T  +  (1+eps[l]) * x[b,i,d] )
// tiles: b(16) x i-tile(16) x d-tile(4) = 1024 waves, 8 WMMA each (K=256).
// ---------------------------------------------------------------------------
__global__ void k_adj_gemm(const _Float16* __restrict__ adj16,
                           const _Float16* __restrict__ msgT,
                           const float* __restrict__ x,
                           const float* __restrict__ eps, int l,
                           _Float16* __restrict__ h16) {
    int lane = threadIdx.x & 31;
    int tile = blockIdx.x * (blockDim.x >> 5) + (threadIdx.x >> 5);
    int tn = tile & 3;
    int ti = (tile >> 2) & 15;
    int b  = tile >> 6;

    const _Float16* A  = adj16 + (size_t)b * GN * GN + (ti * 16) * GN;
    const _Float16* Bt = msgT  + (size_t)b * GD * GN + (tn * 16) * GN;
    int m    = lane & 15;
    int koff = (lane >> 4) << 3;

    v8f acc = {};
    for (int k0 = 0; k0 < GN; k0 += 32) {
        v16h av = ld16(A  + m * GN + k0 + koff);
        v16h bv = ld16(Bt + m * GN + k0 + koff);
        acc = wmma_f16(av, bv, acc);
    }

    float e = 1.0f + eps[l];
    int rbase = ti * 16 + ((lane >> 4) << 3);
    int col   = tn * 16 + (lane & 15);
    size_t base = (size_t)b * GN * GD;
#pragma unroll
    for (int r = 0; r < 8; ++r) {
        size_t idx = base + (size_t)(rbase + r) * GD + col;
        h16[idx] = (_Float16)(acc[r] + e * x[idx]);
    }
}

// ---------------------------------------------------------------------------
// BatchNorm (training mode): per-channel sum / sumsq over all 4096 rows.
// One block per channel.
// ---------------------------------------------------------------------------
__global__ void k_bn_stats(const float* __restrict__ z,
                           float* __restrict__ stats, int C) {
    __shared__ float ssum[256];
    __shared__ float ssq[256];
    int c = blockIdx.x;
    float s = 0.0f, q = 0.0f;
    for (int r = threadIdx.x; r < GM; r += 256) {
        float v = z[(size_t)r * C + c];
        s += v; q += v * v;
    }
    ssum[threadIdx.x] = s; ssq[threadIdx.x] = q;
    __syncthreads();
    for (int off = 128; off > 0; off >>= 1) {
        if ((int)threadIdx.x < off) {
            ssum[threadIdx.x] += ssum[threadIdx.x + off];
            ssq[threadIdx.x]  += ssq[threadIdx.x + off];
        }
        __syncthreads();
    }
    if (threadIdx.x == 0) { stats[c] = ssum[0]; stats[C + c] = ssq[0]; }
}

// BN apply + ReLU -> f16 (feeds the second MLP GEMM)
__global__ void k_bn_relu_f16(const float* __restrict__ z,
                              const float* __restrict__ stats,
                              const float* __restrict__ gamma,
                              const float* __restrict__ beta,
                              _Float16* __restrict__ out, int C) {
    const float invM = 1.0f / (float)GM;
    int n = GM * C;
    for (int i = blockIdx.x * blockDim.x + threadIdx.x; i < n;
         i += gridDim.x * blockDim.x) {
        int c = i % C;
        float mean = stats[c] * invM;
        float var  = stats[C + c] * invM - mean * mean;
        float sc   = rsqrtf(var + 1e-5f) * gamma[c];
        float v    = (z[i] - mean) * sc + beta[c];
        out[i] = (_Float16)fmaxf(v, 0.0f);
    }
}

// BN apply (+optional ReLU) -> f32 (next-layer x, or final output)
__global__ void k_bn_f32(const float* __restrict__ z,
                         const float* __restrict__ stats,
                         const float* __restrict__ gamma,
                         const float* __restrict__ beta,
                         float* __restrict__ out, int C, int doRelu) {
    const float invM = 1.0f / (float)GM;
    int n = GM * C;
    for (int i = blockIdx.x * blockDim.x + threadIdx.x; i < n;
         i += gridDim.x * blockDim.x) {
        int c = i % C;
        float mean = stats[c] * invM;
        float var  = stats[C + c] * invM - mean * mean;
        float sc   = rsqrtf(var + 1e-5f) * gamma[c];
        float v    = (z[i] - mean) * sc + beta[c];
        out[i] = doRelu ? fmaxf(v, 0.0f) : v;
    }
}

// ---------------------------------------------------------------------------
extern "C" void kernel_launch(void* const* d_in, const int* in_sizes, int n_in,
                              void* d_out, int out_size, void* d_ws, size_t ws_size,
                              hipStream_t stream) {
    (void)in_sizes; (void)n_in; (void)out_size; (void)ws_size;

    const float* fea   = (const float*)d_in[0];   // [16,256,32]
    const float* adj   = (const float*)d_in[1];   // [16,256,256]
    const float* encW  = (const float*)d_in[2];   // [32,64]
    const float* encb  = (const float*)d_in[3];   // [64]
    const float* bondW = (const float*)d_in[4];   // [3,64]
    const float* eps   = (const float*)d_in[5];   // [3]
    const float* W1    = (const float*)d_in[6];   // [3,64,128]
    const float* b1    = (const float*)d_in[7];   // [3,128]
    const float* g1    = (const float*)d_in[8];   // [3,128]
    const float* be1   = (const float*)d_in[9];   // [3,128]
    const float* W2    = (const float*)d_in[10];  // [3,128,64]
    const float* b2    = (const float*)d_in[11];  // [3,64]
    const float* gbn   = (const float*)d_in[12];  // [3,64]
    const float* bbn   = (const float*)d_in[13];  // [3,64]
    float* out = (float*)d_out;                   // [16,256,64]

    // ---- workspace layout (256B aligned slices) ----
    char* w = (char*)d_ws;
    size_t off = 0;
    auto alloc = [&](size_t bytes) -> void* {
        void* p = w + off;
        off = (off + bytes + 255) & ~(size_t)255;
        return p;
    };
    _Float16* adj16  = (_Float16*)alloc((size_t)GB * GN * GN * 2); // 2 MB
    _Float16* fea16  = (_Float16*)alloc((size_t)GM * 32 * 2);
    _Float16* encWt  = (_Float16*)alloc((size_t)64 * 32 * 2);      // [64,32]
    _Float16* W1t    = (_Float16*)alloc((size_t)3 * 128 * 64 * 2); // [l][128,64]
    _Float16* W2t    = (_Float16*)alloc((size_t)3 * 64 * 128 * 2); // [l][64,128]
    float*    x      = (float*)   alloc((size_t)GM * GD * 4);
    _Float16* msgT   = (_Float16*)alloc((size_t)GB * GD * GN * 2); // [b][d][j]
    _Float16* h16    = (_Float16*)alloc((size_t)GM * GD * 2);
    float*    z1     = (float*)   alloc((size_t)GM * 128 * 4);
    _Float16* z1r    = (_Float16*)alloc((size_t)GM * 128 * 2);
    float*    z2     = (float*)   alloc((size_t)GM * GD * 4);
    float*    stats  = (float*)   alloc(2 * 128 * 4);

    // ---- one-time (per-call) conversions ----
    k_f32_to_f16<<<256, 256, 0, stream>>>(adj, adj16, GB * GN * GN, 1);
    k_f32_to_f16<<<128, 256, 0, stream>>>(fea, fea16, GM * 32, 0);
    k_transpose_f16<<<8, 256, 0, stream>>>(encW, encWt, 32, 64);
    for (int l = 0; l < 3; ++l) {
        k_transpose_f16<<<32, 256, 0, stream>>>(W1 + l * 64 * 128,
                                                W1t + l * 128 * 64, 64, 128);
        k_transpose_f16<<<32, 256, 0, stream>>>(W2 + l * 128 * 64,
                                                W2t + l * 64 * 128, 128, 64);
    }

    // ---- encoder: x = fea @ enc_W + enc_b  (tiles 256x4 = 1024 waves) ----
    k_gemm_f16<<<256, 128, 0, stream>>>(fea16, encWt, encb, x, 32, GD);

    // ---- 3 GIN layers ----
    for (int l = 0; l < 3; ++l) {
        // message: relu(x + bond) stored column-major per batch
        k_message<<<1024, 256, 0, stream>>>(x, bondW, l, msgT);
        // agg + residual: h16 = adj @ msg + (1+eps)*x   (1024 waves, 8 WMMA)
        k_adj_gemm<<<256, 128, 0, stream>>>(adj16, msgT, x, eps, l, h16);
        // MLP linear 1: z1 = h16 @ W1 + b1   (tiles 256x8 = 2048 waves)
        k_gemm_f16<<<512, 128, 0, stream>>>(h16, W1t + l * 128 * 64,
                                            b1 + l * 128, z1, GD, 128);
        // BN (train) + ReLU -> f16
        k_bn_stats<<<128, 256, 0, stream>>>(z1, stats, 128);
        k_bn_relu_f16<<<2048, 256, 0, stream>>>(z1, stats, g1 + l * 128,
                                                be1 + l * 128, z1r, 128);
        // MLP linear 2: z2 = z1r @ W2 + b2   (tiles 256x4 = 1024 waves)
        k_gemm_f16<<<256, 128, 0, stream>>>(z1r, W2t + l * 64 * 128,
                                            b2 + l * 64, z2, 128, GD);
        // outer BN; ReLU except last layer; last layer writes d_out
        k_bn_stats<<<64, 256, 0, stream>>>(z2, stats, GD);
        float* dst = (l == 2) ? out : x;
        k_bn_f32<<<1024, 256, 0, stream>>>(z2, stats, gbn + l * 64,
                                           bbn + l * 64, dst, GD,
                                           (l < 2) ? 1 : 0);
    }
}
